// Glm4Attention_52810917872272
// MI455X (gfx1250) — matrix operand order repaired and checked by reading, not compiled, not run on hardware
//
#include <hip/hip_runtime.h>
#include <hip/hip_bf16.h>

#define NUM_HEADS 32
#define NUM_KV    2
#define HEAD_DIM  128
#define ROT       64
#define HIDDEN    4096
#define SEQ       2048
#define QKV_OUT   ((NUM_HEADS + 2 * NUM_KV) * HEAD_DIM)   // 4608

typedef __bf16 bf16_t;
typedef __attribute__((ext_vector_type(16))) __bf16 v16bf;
typedef __attribute__((ext_vector_type(8)))  __bf16 v8bf;
typedef __attribute__((ext_vector_type(8)))  float  v8f;
typedef __attribute__((ext_vector_type(4)))  unsigned int v4u;
typedef __attribute__((ext_vector_type(8)))  int v8i;
typedef __attribute__((ext_vector_type(4)))  int v4i;

union ATile { v16bf v; v8bf h[2]; };

#if __has_builtin(__builtin_amdgcn_tensor_load_to_lds)
#define USE_TDM 1
#else
#define USE_TDM 0
#endif

// LDS row strides (elements) after TDM padding -> conflict-free on 64 banks
#define BROW 40    // 32 data + 8 pad   (16 DW + 4 DW)
#define KROW 136   // 128 data + 8 pad  (64 DW + 4 DW)

__device__ __forceinline__ v8f wmma_bf16(v16bf a, v16bf b, v8f c) {
  return __builtin_amdgcn_wmma_f32_16x16x32_bf16(false, a, false, b, (short)0, c,
                                                 false, false);
}

#if USE_TDM
__device__ __forceinline__ unsigned int lds_offset_of(const void* p) {
  return (unsigned int)(unsigned long long)
      (__attribute__((address_space(3))) const char*)p;
}

// 2D TDM load of a (tile_n rows x tile_k elements) bf16 tile, global row
// stride = stride_elems, with LDS padding (pad codes per D# group1 encoding).
__device__ __forceinline__ void tdm_load_2d(unsigned int lds_addr,
                                            const void* gptr,
                                            unsigned int tile_k,
                                            unsigned int tile_n,
                                            unsigned int tensor_k,
                                            unsigned int tensor_n,
                                            unsigned int stride_elems,
                                            unsigned int pad_int_code,
                                            unsigned int pad_amt_code) {
  unsigned long long ga = (unsigned long long)gptr;
  v4u g0;
  g0[0] = 1u;                                   // count=1, user descriptor
  g0[1] = lds_addr;                             // LDS byte address
  g0[2] = (unsigned int)ga;                     // global_addr[31:0]
  g0[3] = (unsigned int)((ga >> 32) & 0x1FFFFFFu) | (2u << 30);  // type=2
  v8i g1;
  // data_size=1 (2B) | pad_enable | pad_interval | pad_amount
  g1[0] = (int)((1u << 16) | (1u << 20) | (pad_int_code << 22) |
                (pad_amt_code << 25));
  g1[1] = (int)((tensor_k & 0xFFFFu) << 16);                       // dim0 lo16
  g1[2] = (int)(((tensor_k >> 16) & 0xFFFFu) | ((tensor_n & 0xFFFFu) << 16));
  g1[3] = (int)(((tensor_n >> 16) & 0xFFFFu) | ((tile_k & 0xFFFFu) << 16));
  g1[4] = (int)(tile_n & 0xFFFFu);              // tile_dim1 ; tile_dim2 = 0
  g1[5] = (int)stride_elems;                    // tensor_dim0_stride[31:0]
  g1[6] = 0;
  g1[7] = 0;
  v4i g2 = {0, 0, 0, 0};
  v4i g3 = {0, 0, 0, 0};
#if __clang_major__ >= 23
  v8i g4 = {0, 0, 0, 0, 0, 0, 0, 0};
  __builtin_amdgcn_tensor_load_to_lds(g0, g1, g2, g3, g4, 0);
#else
  __builtin_amdgcn_tensor_load_to_lds(g0, g1, g2, g3, 0);
#endif
}
#endif  // USE_TDM

// ---------------- f32 -> bf16 conversion (same layout) ----------------
__global__ void k_cvt(const float* __restrict__ src, bf16_t* __restrict__ dst, int n) {
  int i = blockIdx.x * blockDim.x + threadIdx.x;
  if (i < n) dst[i] = (bf16_t)src[i];
}

// ---------------- f32 -> bf16 transpose: dst[n*K+k] = src[k*N+n] ------
__global__ void k_cvt_T(const float* __restrict__ src, bf16_t* __restrict__ dst,
                        int K, int N) {
  int i = blockIdx.x * blockDim.x + threadIdx.x;
  if (i < K * N) {
    int k = i / N;
    int n = i - k * N;
    dst[(size_t)n * K + k] = (bf16_t)src[i];
  }
}

// ---------------- bf16 GEMM: C[M,N] = A[M,K] * BT[N,K]^T (+bias) ------
__global__ void __launch_bounds__(128) k_gemm(const bf16_t* __restrict__ A,
                                              const bf16_t* __restrict__ BT,
                                              const float* __restrict__ bias,
                                              float* __restrict__ C,
                                              int Kdim, int Ncols) {
  __shared__ bf16_t bbuf[2][64 * BROW];
  const int tid   = threadIdx.x;
  const int lane  = tid & 31;
  const int wave  = tid >> 5;
  const int mbase = (blockIdx.x * 4 + wave) * 16;
  const int nbase = blockIdx.y * 64;
  const int l15   = lane & 15;
  const int c0    = (lane < 16) ? 0 : 16;
  const int koff  = (lane < 16) ? 0 : 16;
  const bf16_t* arow = A + (size_t)(mbase + l15) * Kdim;
  const int srow  = tid >> 1;
  const int shalf = tid & 1;
  const int iters = Kdim / 32;

#if USE_TDM
  if (wave == 0) {
    tdm_load_2d(lds_offset_of(&bbuf[0][0]), BT + (size_t)nbase * Kdim,
                32u, 64u, (unsigned)Kdim, (unsigned)Ncols, (unsigned)Kdim, 3u, 3u);
    __builtin_amdgcn_s_wait_tensorcnt(0);
  }
#else
  {
    v16bf r = *(const v16bf*)(BT + (size_t)(nbase + srow) * Kdim + shalf * 16);
    *(v16bf*)&bbuf[0][srow * BROW + shalf * 16] = r;
  }
#endif
  __syncthreads();

  ATile a_cur, a_nxt;
  a_cur.h[0] = *(const v8bf*)(arow + c0);
  a_cur.h[1] = *(const v8bf*)(arow + 8 + c0);

  v8f acc0 = {}, acc1 = {}, acc2 = {}, acc3 = {};
  for (int i = 0; i < iters; ++i) {
    const int k_nxt = (i + 1) * 32;
    const bool has_nxt = (i + 1) < iters;
#if USE_TDM
    if (has_nxt && wave == 0) {
      tdm_load_2d(lds_offset_of(&bbuf[(i + 1) & 1][0]),
                  BT + (size_t)nbase * Kdim + k_nxt,
                  32u, 64u, (unsigned)Kdim, (unsigned)Ncols, (unsigned)Kdim, 3u, 3u);
    }
#else
    v16bf breg = {};
    if (has_nxt)
      breg = *(const v16bf*)(BT + (size_t)(nbase + srow) * Kdim + k_nxt + shalf * 16);
#endif
    if (has_nxt) {
      a_nxt.h[0] = *(const v8bf*)(arow + k_nxt + c0);
      a_nxt.h[1] = *(const v8bf*)(arow + k_nxt + 8 + c0);
    }
    __builtin_prefetch(arow + k_nxt + 96, 0, 1);

    const bf16_t* bb = &bbuf[i & 1][0];
    v16bf b0 = *(const v16bf*)(bb + (size_t)(l15)      * BROW + koff);
    v16bf b1 = *(const v16bf*)(bb + (size_t)(16 + l15) * BROW + koff);
    v16bf b2 = *(const v16bf*)(bb + (size_t)(32 + l15) * BROW + koff);
    v16bf b3 = *(const v16bf*)(bb + (size_t)(48 + l15) * BROW + koff);
    acc0 = wmma_bf16(a_cur.v, b0, acc0);
    acc1 = wmma_bf16(a_cur.v, b1, acc1);
    acc2 = wmma_bf16(a_cur.v, b2, acc2);
    acc3 = wmma_bf16(a_cur.v, b3, acc3);
    a_cur = a_nxt;

#if USE_TDM
    if (has_nxt && wave == 0) __builtin_amdgcn_s_wait_tensorcnt(0);
    __syncthreads();
#else
    __syncthreads();
    if (has_nxt)
      *(v16bf*)&bbuf[(i + 1) & 1][srow * BROW + shalf * 16] = breg;
    __syncthreads();
#endif
  }

  const int mo   = (lane < 16) ? 0 : 8;
  const int ncol = nbase + l15;
  float bb0 = bias ? bias[ncol]      : 0.f;
  float bb1 = bias ? bias[ncol + 16] : 0.f;
  float bb2 = bias ? bias[ncol + 32] : 0.f;
  float bb3 = bias ? bias[ncol + 48] : 0.f;
  for (int r = 0; r < 8; ++r) {
    float* crow = C + (size_t)(mbase + r + mo) * Ncols;
    crow[ncol]      = acc0[r] + bb0;
    crow[ncol + 16] = acc1[r] + bb1;
    crow[ncol + 32] = acc2[r] + bb2;
    crow[ncol + 48] = acc3[r] + bb3;
  }
}

// ---------------- RoPE + scatter QKV into attention layouts ----------
__global__ void k_rope_scatter(const float* __restrict__ qkv,
                               const float* __restrict__ cosb,
                               const float* __restrict__ sinb,
                               bf16_t* __restrict__ qb,
                               bf16_t* __restrict__ kb,
                               bf16_t* __restrict__ vtb) {
  int idx = blockIdx.x * blockDim.x + threadIdx.x;
  if (idx >= SEQ * QKV_OUT) return;
  int s = idx / QKV_OUT;
  int c = idx - s * QKV_OUT;
  float val = qkv[idx];
  if (c < NUM_HEADS * HEAD_DIM) {
    int head = c >> 7, d = c & 127;
    if (d < ROT) {
      int i = d >> 1;
      float cs = cosb[s * ROT + i], sn = sinb[s * ROT + i];
      float other = qkv[(size_t)s * QKV_OUT + (c ^ 1)];
      val = val * cs + ((d & 1) ? other : -other) * sn;
    }
    qb[((size_t)head * SEQ + s) * HEAD_DIM + d] = (bf16_t)val;
  } else if (c < (NUM_HEADS + NUM_KV) * HEAD_DIM) {
    int cc = c - NUM_HEADS * HEAD_DIM;
    int kv = cc >> 7, d = cc & 127;
    if (d < ROT) {
      int i = d >> 1;
      float cs = cosb[s * ROT + i], sn = sinb[s * ROT + i];
      float other = qkv[(size_t)s * QKV_OUT + (c ^ 1)];
      val = val * cs + ((d & 1) ? other : -other) * sn;
    }
    kb[((size_t)kv * SEQ + s) * HEAD_DIM + d] = (bf16_t)val;
  } else {
    int cc = c - (NUM_HEADS + NUM_KV) * HEAD_DIM;
    int kv = cc >> 7, d = cc & 127;
    vtb[((size_t)kv * HEAD_DIM + d) * SEQ + s] = (bf16_t)val;
  }
}

// ---------------- Flash attention ------------------------------------
// Block = 4 waves = 64 consecutive queries of one head. All waves share the
// K/V tiles of each 32-key step, staged into LDS by the TDM (double-buffered,
// overlapped with WMMA). Waves past their causal range skip compute
// (wave-uniform branch; EXEC stays all-ones inside WMMA).
__global__ void __launch_bounds__(128) k_attn(const bf16_t* __restrict__ qb,
                                              const bf16_t* __restrict__ kb,
                                              const bf16_t* __restrict__ vtb,
                                              bf16_t* __restrict__ attnb) {
  __shared__ bf16_t ktile[2][32 * KROW];    // [key][d] rows, padded
  __shared__ bf16_t vtile[2][128 * BROW];   // [d][key] rows, padded
  __shared__ bf16_t pbuf[4][16 * 32];
  const int tid   = threadIdx.x;
  const int lane  = tid & 31;
  const int wave  = tid >> 5;
  const int head  = blockIdx.y;
  const int qt    = blockIdx.x * 4 + wave;
  const int kv    = head / (NUM_HEADS / NUM_KV);
  const int qbase = qt * 16;
  const int l15   = lane & 15;
  const int c0    = (lane < 16) ? 0 : 16;
  const int koff  = (lane < 16) ? 0 : 16;
  const int mo    = (lane < 16) ? 0 : 8;
  const float scale = 0.08838834764831845f;  // 1/sqrt(128)

  const bf16_t* kp = kb + (size_t)kv * SEQ * HEAD_DIM;
  const bf16_t* vp = vtb + (size_t)kv * HEAD_DIM * SEQ;

  // Q A-fragments for this wave's 16 queries (4 K-chunks of 32 over d=128)
  const bf16_t* qrow = qb + ((size_t)head * SEQ + qbase + l15) * HEAD_DIM;
  ATile qa[4];
  for (int t = 0; t < 4; ++t) {
    qa[t].h[0] = *(const v8bf*)(qrow + t * 32 + c0);
    qa[t].h[1] = *(const v8bf*)(qrow + t * 32 + 8 + c0);
  }
  v8f acc[8] = {};
  float rowmax[8], rowsum[8];
  for (int r = 0; r < 8; ++r) { rowmax[r] = -1e30f; rowsum[r] = 0.f; }
  bf16_t* myp = pbuf[wave];

  const int jend_my  = qbase + 16;
  const int nsteps   = 2 * (blockIdx.x + 1);   // block covers keys [0, 64*(bx+1))

  // ---- prologue: stage K/V tiles for jb = 0 ----
#if USE_TDM
  if (wave == 0) {
    tdm_load_2d(lds_offset_of(&ktile[0][0]), kp,
                128u, 32u, 128u, (unsigned)SEQ, 128u, 5u, 3u);
    tdm_load_2d(lds_offset_of(&vtile[0][0]), vp,
                32u, 128u, (unsigned)SEQ, 128u, (unsigned)SEQ, 3u, 3u);
    __builtin_amdgcn_s_wait_tensorcnt(0);
  }
#else
  {
    int krow = tid >> 2, kq = tid & 3;
    *(v16bf*)&ktile[0][krow * KROW + kq * 32] =
        *(const v16bf*)(kp + (size_t)krow * HEAD_DIM + kq * 32);
    *(v16bf*)&ktile[0][krow * KROW + kq * 32 + 16] =
        *(const v16bf*)(kp + (size_t)krow * HEAD_DIM + kq * 32 + 16);
    *(v16bf*)&vtile[0][tid * BROW] = *(const v16bf*)(vp + (size_t)tid * SEQ);
    *(v16bf*)&vtile[0][tid * BROW + 16] =
        *(const v16bf*)(vp + (size_t)tid * SEQ + 16);
  }
#endif
  __syncthreads();

  for (int i = 0; i < nsteps; ++i) {
    const int jb = i * 32;
    const bool has_nxt = (i + 1) < nsteps;
    const int jb_nxt = jb + 32;
#if USE_TDM
    if (has_nxt && wave == 0) {
      tdm_load_2d(lds_offset_of(&ktile[(i + 1) & 1][0]),
                  kp + (size_t)jb_nxt * HEAD_DIM,
                  128u, 32u, 128u, (unsigned)SEQ, 128u, 5u, 3u);
      tdm_load_2d(lds_offset_of(&vtile[(i + 1) & 1][0]), vp + jb_nxt,
                  32u, 128u, (unsigned)SEQ, 128u, (unsigned)SEQ, 3u, 3u);
    }
#else
    v16bf kr0 = {}, kr1 = {}, vr0 = {}, vr1 = {};
    int krow = tid >> 2, kq = tid & 3;
    if (has_nxt) {
      kr0 = *(const v16bf*)(kp + (size_t)(jb_nxt + krow) * HEAD_DIM + kq * 32);
      kr1 = *(const v16bf*)(kp + (size_t)(jb_nxt + krow) * HEAD_DIM + kq * 32 + 16);
      vr0 = *(const v16bf*)(vp + (size_t)tid * SEQ + jb_nxt);
      vr1 = *(const v16bf*)(vp + (size_t)tid * SEQ + jb_nxt + 16);
    }
#endif
    if (jb < jend_my) {  // wave-uniform causal skip
      const bf16_t* kt = &ktile[i & 1][0];
      const bf16_t* vt = &vtile[i & 1][0];
      v8f s0 = {}, s1 = {};
      for (int t = 0; t < 4; ++t) {
        v16bf bk0 = *(const v16bf*)(kt + (size_t)(l15)      * KROW + t * 32 + koff);
        v16bf bk1 = *(const v16bf*)(kt + (size_t)(16 + l15) * KROW + t * 32 + koff);
        s0 = wmma_bf16(qa[t].v, bk0, s0);
        s1 = wmma_bf16(qa[t].v, bk1, s1);
      }
      const int n0 = jb + l15, n1 = n0 + 16;
      for (int r = 0; r < 8; ++r) {
        const int qi = qbase + r + mo;
        float x0 = (n0 <= qi) ? s0[r] * scale : -1e30f;
        float x1 = (n1 <= qi) ? s1[r] * scale : -1e30f;
        float mx = fmaxf(x0, x1);
        for (int off = 1; off < 16; off <<= 1) mx = fmaxf(mx, __shfl_xor(mx, off, 32));
        float Mnew  = fmaxf(rowmax[r], mx);
        float alpha = __expf(rowmax[r] - Mnew);
        rowmax[r] = Mnew;
        float p0 = __expf(x0 - Mnew);
        float p1 = __expf(x1 - Mnew);
        float rs = p0 + p1;
        for (int off = 1; off < 16; off <<= 1) rs += __shfl_xor(rs, off, 32);
        rowsum[r] = rowsum[r] * alpha + rs;
        for (int t = 0; t < 8; ++t) acc[t][r] *= alpha;
        myp[(r + mo) * 32 + l15]      = (bf16_t)p0;
        myp[(r + mo) * 32 + l15 + 16] = (bf16_t)p1;
      }
      asm volatile("s_wait_dscnt 0" ::: "memory");
      ATile pa;
      pa.h[0] = *(const v8bf*)(myp + l15 * 32 + c0);
      pa.h[1] = *(const v8bf*)(myp + l15 * 32 + 8 + c0);
      asm volatile("s_wait_dscnt 0" ::: "memory");
      for (int t = 0; t < 8; ++t) {
        v16bf bv = *(const v16bf*)(vt + (size_t)(t * 16 + l15) * BROW + koff);
        acc[t] = wmma_bf16(pa.v, bv, acc[t]);
      }
    }
#if USE_TDM
    if (has_nxt && wave == 0) __builtin_amdgcn_s_wait_tensorcnt(0);
    __syncthreads();
#else
    __syncthreads();
    if (has_nxt) {
      *(v16bf*)&ktile[(i + 1) & 1][krow * KROW + kq * 32]      = kr0;
      *(v16bf*)&ktile[(i + 1) & 1][krow * KROW + kq * 32 + 16] = kr1;
      *(v16bf*)&vtile[(i + 1) & 1][tid * BROW]      = vr0;
      *(v16bf*)&vtile[(i + 1) & 1][tid * BROW + 16] = vr1;
    }
    __syncthreads();
#endif
  }

  for (int r = 0; r < 8; ++r) {
    float inv = 1.0f / rowsum[r];
    bf16_t* orow = attnb + (size_t)(qbase + r + mo) * (NUM_HEADS * HEAD_DIM)
                 + head * HEAD_DIM + l15;
    for (int t = 0; t < 8; ++t) orow[t * 16] = (bf16_t)(acc[t][r] * inv);
  }
}

extern "C" void kernel_launch(void* const* d_in, const int* in_sizes, int n_in,
                              void* d_out, int out_size, void* d_ws, size_t ws_size,
                              hipStream_t stream) {
  const float* hidden = (const float*)d_in[0];
  const float* cosb   = (const float*)d_in[1];
  const float* sinb   = (const float*)d_in[2];
  const float* w_qkv  = (const float*)d_in[3];
  const float* b_qkv  = (const float*)d_in[4];
  const float* w_o    = (const float*)d_in[5];

  char* ws = (char*)d_ws;
  size_t off = 0;
  bf16_t* hsb   = (bf16_t*)(ws + off); off += (size_t)SEQ * HIDDEN * 2;
  bf16_t* wqkvT = (bf16_t*)(ws + off); off += (size_t)QKV_OUT * HIDDEN * 2;
  bf16_t* woT   = (bf16_t*)(ws + off); off += (size_t)HIDDEN * HIDDEN * 2;
  float*  qkvf  = (float*) (ws + off); off += (size_t)SEQ * QKV_OUT * 4;
  bf16_t* qbuf  = (bf16_t*)(ws + off); off += (size_t)NUM_HEADS * SEQ * HEAD_DIM * 2;
  bf16_t* kbuf  = (bf16_t*)(ws + off); off += (size_t)NUM_KV * SEQ * HEAD_DIM * 2;
  bf16_t* vtbuf = (bf16_t*)(ws + off); off += (size_t)NUM_KV * SEQ * HEAD_DIM * 2;
  bf16_t* attnb = (bf16_t*)(ws + off); off += (size_t)SEQ * NUM_HEADS * HEAD_DIM * 2;

  int n1 = SEQ * HIDDEN;
  k_cvt<<<(n1 + 255) / 256, 256, 0, stream>>>(hidden, hsb, n1);
  int n2 = HIDDEN * QKV_OUT;
  k_cvt_T<<<(n2 + 255) / 256, 256, 0, stream>>>(w_qkv, wqkvT, HIDDEN, QKV_OUT);
  int n3 = HIDDEN * HIDDEN;
  k_cvt_T<<<(n3 + 255) / 256, 256, 0, stream>>>(w_o, woT, HIDDEN, HIDDEN);

  dim3 g1(SEQ / 64, QKV_OUT / 64);
  k_gemm<<<g1, 128, 0, stream>>>(hsb, wqkvT, b_qkv, qkvf, HIDDEN, QKV_OUT);

  int n4 = SEQ * QKV_OUT;
  k_rope_scatter<<<(n4 + 255) / 256, 256, 0, stream>>>(qkvf, cosb, sinb,
                                                       qbuf, kbuf, vtbuf);

  dim3 g2(SEQ / 64, NUM_HEADS);
  k_attn<<<g2, 128, 0, stream>>>(qbuf, kbuf, vtbuf, attnb);

  dim3 g3(SEQ / 64, HIDDEN / 64);
  k_gemm<<<g3, 128, 0, stream>>>(attnb, woT, nullptr, (float*)d_out,
                                 HIDDEN, HIDDEN);
}
